// PostJointNet_50414326121239
// MI455X (gfx1250) — matrix operand order repaired and checked by reading, not compiled, not run on hardware
//
#include <hip/hip_runtime.h>
#include <math.h>

constexpr int HID  = 384;   // hidden size
constexpr int EPAD = 776;   // 768 + 8 halves pad -> conflict-free A-frag reads
constexpr int HPAD = 392;   // 384 + 8 halves pad
constexpr int WAVES = 4;    // waves per block, 16 edges per wave

typedef _Float16 h4   __attribute__((ext_vector_type(4)));
typedef _Float16 h8   __attribute__((ext_vector_type(8)));
typedef _Float16 v16h __attribute__((ext_vector_type(16)));
typedef float    v8f  __attribute__((ext_vector_type(8)));

// ---------------- fp32 -> f16 weight conversion ----------------
__global__ void cvt_f16(const float* __restrict__ s, _Float16* __restrict__ d, int n) {
    int i = blockIdx.x * blockDim.x + threadIdx.x;
    if (i < n) d[i] = (_Float16)s[i];
}

__device__ __forceinline__ v16h cat16(h8 lo, h8 hi) {
    return __builtin_shufflevector(lo, hi, 0,1,2,3,4,5,6,7,8,9,10,11,12,13,14,15);
}

// Branchless ELU: for v>0 -> v + (exp(0)-1) = v ; for v<=0 -> 0 + exp(v)-1.
// __expf = single v_exp_f32 (arg is <= 0 so result in (0,1], no range issues);
// keeps EXEC all-ones (no divergence around the WMMA pipeline).
__device__ __forceinline__ float elu(float v) {
    return fmaxf(v, 0.f) + (__expf(fminf(v, 0.f)) - 1.f);
}

// One MLP layer for a 16-edge tile:
//   out[16][384] = ELU( in[16][KDIM] * W^T + bias ), f16 out to LDS
template <int KDIM>
__device__ void mlp_layer(const _Float16* __restrict__ in_tile, int in_pad,
                          const _Float16* __restrict__ Wh,     // [384][KDIM] row-major f16
                          const float*    __restrict__ bias,
                          _Float16*       __restrict__ out_tile, int out_pad,
                          int lane) {
    const int col = lane & 15;   // A row (edge), B column (out channel), C column
    const int hi  = lane >> 4;   // half-wave selector
    for (int nt = 0; nt < HID / 16; ++nt) {
        const float bv = bias[nt * 16 + col];
        v8f acc = {bv, bv, bv, bv, bv, bv, bv, bv};
        const _Float16* wrow = Wh + (size_t)(nt * 16 + col) * KDIM;
#pragma unroll 4
        for (int kb = 0; kb < KDIM / 32; ++kb) {
            const int k0 = kb * 32;
            // A frag (16x32 f16): lanes 0-15 hold K {0..7,16..23}, lanes 16-31 K {8..15,24..31}
            const _Float16* ap = in_tile + col * in_pad + k0 + hi * 8;
            v16h a = cat16(*(const h8*)ap, *(const h8*)(ap + 16));
            // B frag (32x16 f16): lane n holds column n = W row n; lanes 0-15 K {0..7,16..23}, 16-31 K {8..15,24..31}
            const h8* wp = (const h8*)(wrow + k0 + hi * 8);
            v16h b = cat16(wp[0], wp[2]);
            acc = __builtin_amdgcn_wmma_f32_16x16x32_f16(
                false, a, false, b, (short)0, acc, false, false);
        }
        // ELU on f32 accumulator, store f16 tile (C/D layout: VGPR r -> M = r + 8*hi)
#pragma unroll
        for (int r = 0; r < 8; ++r) {
            out_tile[(r + 8 * hi) * out_pad + nt * 16 + col] = (_Float16)elu(acc[r]);
        }
    }
}

__global__ void __launch_bounds__(WAVES * 32)
edge_mlp(const float* __restrict__ x1, const float* __restrict__ x2,
         const int*   __restrict__ ei,
         const _Float16* __restrict__ W1h, const float* __restrict__ b1,
         const _Float16* __restrict__ W2h, const float* __restrict__ b2,
         const float* __restrict__ W3,  const float* __restrict__ b3,
         float* __restrict__ out, int E, int N1) {
    __shared__ _Float16 smem[WAVES * (16 * EPAD + 16 * HPAD)];
    const int lane = threadIdx.x & 31;
    const int w    = threadIdx.x >> 5;
    _Float16* e_tile = smem + w * (16 * EPAD + 16 * HPAD);
    _Float16* h_tile = e_tile + 16 * EPAD;
    const int eb = blockIdx.x * (WAVES * 16) + w * 16;

    // ---- gather: 32 lanes fetch both endpoint rows of each edge, convert to f16 in LDS ----
    for (int i = 0; i < 16; ++i) {
        const int e = eb + i;
        if (e >= E) break;                       // uniform per wave
        const int si = ei[e];
        const int di = ei[E + e];
        const float* ps = (si < N1) ? (x1 + (size_t)si * HID) : (x2 + (size_t)(si - N1) * HID);
        const float* pd = (di < N1) ? (x1 + (size_t)di * HID) : (x2 + (size_t)(di - N1) * HID);
        _Float16* row = e_tile + i * EPAD;
#pragma unroll
        for (int j = 0; j < 3; ++j) {
            const int q = j * 32 + lane;         // float4 index, 96 per row
            float4 vs = ((const float4*)ps)[q];
            float4 vd = ((const float4*)pd)[q];
            h4 hs = {(_Float16)vs.x, (_Float16)vs.y, (_Float16)vs.z, (_Float16)vs.w};
            h4 hd = {(_Float16)vd.x, (_Float16)vd.y, (_Float16)vd.z, (_Float16)vd.w};
            *(h4*)(row + q * 4)       = hs;
            *(h4*)(row + HID + q * 4) = hd;
        }
    }

    // ---- layer 1: [16x768] x [768x384] ; layer 2: [16x384] x [384x384] (reuses e_tile) ----
    mlp_layer<2 * HID>(e_tile, EPAD, W1h, b1, h_tile, HPAD, lane);
    mlp_layer<HID>(h_tile, HPAD, W2h, b2, e_tile, EPAD, lane);

    // ---- layer 3: logits[16] = h2 . W3 + b3 ; two lanes per edge ----
    const int edge = lane & 15, half = lane >> 4;
    float sum = 0.f;
    const _Float16* hr = e_tile + edge * EPAD + half * 192;
    const float4*   w4 = (const float4*)(W3 + half * 192);
#pragma unroll 4
    for (int c8 = 0; c8 < 24; ++c8) {
        h8 hv = *(const h8*)(hr + c8 * 8);
        float4 wa = w4[c8 * 2], wb = w4[c8 * 2 + 1];
        sum += (float)hv[0] * wa.x + (float)hv[1] * wa.y +
               (float)hv[2] * wa.z + (float)hv[3] * wa.w +
               (float)hv[4] * wb.x + (float)hv[5] * wb.y +
               (float)hv[6] * wb.z + (float)hv[7] * wb.w;
    }
    sum += __shfl_xor(sum, 16, 32);
    if (half == 0 && eb + edge < E) out[eb + edge] = sum + b3[0];
}

extern "C" void kernel_launch(void* const* d_in, const int* in_sizes, int n_in,
                              void* d_out, int out_size, void* d_ws, size_t ws_size,
                              hipStream_t stream) {
    const float* x1 = (const float*)d_in[0];
    const float* x2 = (const float*)d_in[1];
    const int*   ei = (const int*)d_in[2];
    const float* W1 = (const float*)d_in[3];
    const float* b1 = (const float*)d_in[4];
    const float* W2 = (const float*)d_in[5];
    const float* b2 = (const float*)d_in[6];
    const float* W3 = (const float*)d_in[7];
    const float* b3 = (const float*)d_in[8];
    float* out = (float*)d_out;

    const int E  = in_sizes[2] / 2;
    const int N1 = in_sizes[0] / HID;

    _Float16* W1h = (_Float16*)d_ws;                 // 384*768 halves
    _Float16* W2h = W1h + HID * 2 * HID;             // 384*384 halves

    const int n1 = HID * 2 * HID;
    cvt_f16<<<(n1 + 255) / 256, 256, 0, stream>>>(W1, W1h, n1);
    const int n2 = HID * HID;
    cvt_f16<<<(n2 + 255) / 256, 256, 0, stream>>>(W2, W2h, n2);

    const int blocks = (E + WAVES * 16 - 1) / (WAVES * 16);
    edge_mlp<<<blocks, WAVES * 32, 0, stream>>>(x1, x2, ei, W1h, b1, W2h, b2,
                                                W3, b3, out, E, N1);
    (void)n_in; (void)out_size; (void)ws_size;
}
